// BasisConvLayer_39161511805168
// MI455X (gfx1250) — matrix operand order repaired and checked by reading, not compiled, not run on hardware
//
#include <hip/hip_runtime.h>
#include <stdint.h>

// ---------------------------------------------------------------------------
// BasisConvLayer:  out[row[e]] += sum_k bx[u]*by[v] * (x[col[e]] @ W[u,v])
// Reformulated as GEMM:  U[E,256] @ Wflat[256,16], U[e, k*16+f] = t_k * x_j[f]
// BF16 WMMA (16x16x32, f32 acc). A fragments built with v_pk_mul_bf16 using
// op_sel broadcasts of packed basis coefficients (v_cvt_pk_bf16_f32).
// One wave per 16-edge tile; scatter via global_atomic_add_f32.
// ---------------------------------------------------------------------------

typedef __attribute__((ext_vector_type(16))) __bf16   v16bf;
typedef __attribute__((ext_vector_type(8)))  float    v8f;
typedef __attribute__((ext_vector_type(8)))  uint32_t v8u;

// two f32 -> packed bf16 dword (lo = a, hi = b), single instruction
__device__ __forceinline__ uint32_t cvt_pk_bf16(float a, float b) {
  uint32_t d;
  asm("v_cvt_pk_bf16_f32 %0, %1, %2" : "=v"(d) : "v"(a), "v"(b));
  return d;
}
// d.{lo,hi} = t.lo * x.{lo,hi}   (broadcast low half of src0)
__device__ __forceinline__ uint32_t pk_mul_bf16_blo(uint32_t t, uint32_t x) {
  uint32_t d;
  asm("v_pk_mul_bf16 %0, %1, %2 op_sel:[0,0] op_sel_hi:[0,1]"
      : "=v"(d) : "v"(t), "v"(x));
  return d;
}
// d.{lo,hi} = t.hi * x.{lo,hi}   (broadcast high half of src0)
__device__ __forceinline__ uint32_t pk_mul_bf16_bhi(uint32_t t, uint32_t x) {
  uint32_t d;
  asm("v_pk_mul_bf16 %0, %1, %2 op_sel:[1,0] op_sel_hi:[1,1]"
      : "=v"(d) : "v"(t), "v"(x));
  return d;
}
__device__ __forceinline__ void gatomic_add_f32(float* p, float v) {
  // fire-and-forget f32 atomic add (no return -> STOREcnt; s_endpgm waits idle)
  asm volatile("global_atomic_add_f32 %0, %1, off"
               :: "v"((uint64_t)(uintptr_t)p), "v"(v)
               : "memory");
}

__global__ void zero_f32(float* __restrict__ p, int n) {
  int i = blockIdx.x * blockDim.x + threadIdx.x;
  if (i < n) p[i] = 0.0f;
}

__global__ __launch_bounds__(256)
void basisconv_wmma(const float* __restrict__ x,       // [N,16]
                    const int*   __restrict__ row,     // [E] dest nodes
                    const int*   __restrict__ col,     // [E] src nodes
                    const float* __restrict__ attr,    // [E,2]
                    const float* __restrict__ w,       // [256,16] flat
                    float*       __restrict__ out,     // [N,16]
                    int nEdges, int nTiles)
{
  const int lane = threadIdx.x & 31;
  const int g    = lane >> 4;   // half-wave group (feature half / K half)
  const int n    = lane & 15;   // output column owned in B/C/D fragments

  // ---- Preload B fragments (Wflat as bf16), shared by all tiles ----------
  // 16-bit B 32x16 layout: dword v holds K = 8*g + 2v + 8*(v>=4), +1 (packed), N=lane%16
  v16bf B[8];
#pragma unroll
  for (int c = 0; c < 8; ++c) {
    v8u b;
#pragma unroll
    for (int v = 0; v < 8; ++v) {
      int K0 = c * 32 + g * 8 + 2 * v + (v >> 2) * 8;
      b[v] = cvt_pk_bf16(w[K0 * 16 + n], w[(K0 + 1) * 16 + n]);
    }
    B[c] = __builtin_bit_cast(v16bf, b);
  }

  const int waveId = blockIdx.x * (blockDim.x >> 5) + (threadIdx.x >> 5);
  const int nWaves = gridDim.x * (blockDim.x >> 5);

  for (int tile = waveId; tile < nTiles; tile += nWaves) {
    const int e0    = tile * 16;
    const int eMine = e0 + (lane & 15);       // edge this lane's A-row covers
    const bool ok   = eMine < nEdges;
    const bool full = (e0 + 16) <= nEdges;

    // ---- Per-edge basis coefficients (hat basis, 4 terms on [-1,1]) ------
    int   j  = ok ? col[eMine]          : 0;
    float dx = ok ? attr[2 * eMine]     : 9.0f;   // OOB -> all bases 0
    float dy = ok ? attr[2 * eMine + 1] : 9.0f;
    float bx[4], by[4];
#pragma unroll
    for (int u = 0; u < 4; ++u) {
      float cu = -1.0f + (2.0f / 3.0f) * (float)u;
      float vx = 1.0f - fabsf(dx - cu) * 1.5f;   // 1/dx_spacing = 1.5
      float vy = 1.0f - fabsf(dy - cu) * 1.5f;
      bx[u] = vx > 0.0f ? vx : 0.0f;
      by[u] = vy > 0.0f ? vy : 0.0f;
    }
    // 8 packed bf16 pairs: tpair[c] = { t_{2c} , t_{2c+1} },  t_{u*4+v}=bx*by
    uint32_t tpair[8];
#pragma unroll
    for (int u = 0; u < 4; ++u) {
      tpair[2 * u]     = cvt_pk_bf16(bx[u] * by[0], bx[u] * by[1]);
      tpair[2 * u + 1] = cvt_pk_bf16(bx[u] * by[2], bx[u] * by[3]);
    }

    // ---- Gather this lane's 8 source features (32B, L2-resident) ---------
    const float* xp = x + (size_t)j * 16 + g * 8;
    float4 lo = *(const float4*)(xp);
    float4 hi = *(const float4*)(xp + 4);
    uint32_t xpk[4] = { cvt_pk_bf16(lo.x, lo.y), cvt_pk_bf16(lo.z, lo.w),
                        cvt_pk_bf16(hi.x, hi.y), cvt_pk_bf16(hi.z, hi.w) };

    // ---- 8 K-chunks of 32: A[e, K] = t_{K/16} * x_j[K%16] ----------------
    // A fragment (lane group g): dword v   = t_{2c}   * x[2v..2v+1]
    //                            dword 4+v = t_{2c+1} * x[2v..2v+1]
    v8f acc = {};
#pragma unroll
    for (int c = 0; c < 8; ++c) {
      v8u au;
#pragma unroll
      for (int v = 0; v < 4; ++v) {
        au[v]     = pk_mul_bf16_blo(tpair[c], xpk[v]);
        au[4 + v] = pk_mul_bf16_bhi(tpair[c], xpk[v]);
      }
      v16bf a = __builtin_bit_cast(v16bf, au);
      acc = __builtin_amdgcn_wmma_f32_16x16x32_bf16(
          /*neg_a=*/false, a, /*neg_b=*/false, B[c],
          /*c_mod=*/(short)0, acc, /*reuse_a=*/false, /*reuse_b=*/false);
    }

    // ---- Scatter: lane holds D[M = r+8g][n]; edge = e0 + r + 8g ----------
    if (full) {
      const int4* rp = (const int4*)(row + e0 + 8 * g);
      int4 r0 = rp[0];
      int4 r1 = rp[1];
      int ridx[8] = {r0.x, r0.y, r0.z, r0.w, r1.x, r1.y, r1.z, r1.w};
#pragma unroll
      for (int r = 0; r < 8; ++r)
        gatomic_add_f32(&out[(size_t)ridx[r] * 16 + n], acc[r]);
    } else {
#pragma unroll
      for (int r = 0; r < 8; ++r) {
        int e = e0 + 8 * g + r;
        if (e < nEdges)
          gatomic_add_f32(&out[(size_t)row[e] * 16 + n], acc[r]);
      }
    }
  }
}

extern "C" void kernel_launch(void* const* d_in, const int* in_sizes, int n_in,
                              void* d_out, int out_size, void* d_ws, size_t ws_size,
                              hipStream_t stream) {
  const float* x    = (const float*)d_in[0];            // [N,16]
  const int*   eidx = (const int*)d_in[1];              // [2,E]
  const float* attr = (const float*)d_in[2];            // [E,2]
  const float* w    = (const float*)d_in[3];            // [4,4,16,16] = [256,16]
  float* out = (float*)d_out;

  const int nEdges = in_sizes[2] / 2;
  const int nTiles = (nEdges + 15) / 16;
  const int* rowp = eidx;
  const int* colp = eidx + nEdges;

  // zero the (poisoned) accumulator buffer
  zero_f32<<<(out_size + 255) / 256, 256, 0, stream>>>(out, out_size);

  dim3 block(256);
  dim3 grid(1024);   // 8192 waves, ~12 tiles each for E=1.6M
  basisconv_wmma<<<grid, block, 0, stream>>>(x, rowp, colp, attr, w, out,
                                             nEdges, nTiles);
}
